// GPyTorchRBFKernelFFN_61186104098901
// MI455X (gfx1250) — compile-verified, matrix operand order checked by reading, not run on hardware
//
#include <hip/hip_runtime.h>

// Problem constants (B=4, L=8192, D=NK=512)
#define BL    32768
#define DDIM  512
#define MBLK  64          // rows per workgroup
#define NWG   (BL / MBLK) // 512 workgroups

typedef __attribute__((ext_vector_type(16))) __bf16 v16bf;
typedef __attribute__((ext_vector_type(8)))  __bf16 v8bf;
typedef __attribute__((ext_vector_type(8)))  float  v8f;
typedef __attribute__((ext_vector_type(4)))  int    v4i;

#define AS1 __attribute__((address_space(1)))
#define AS3 __attribute__((address_space(3)))

#if defined(__has_builtin)
#  if __has_builtin(__builtin_amdgcn_global_load_async_to_lds_b128)
#    define HAVE_ASYNC 1
#  endif
#endif
#ifndef HAVE_ASYNC
#  define HAVE_ASYNC 0
#endif

__device__ inline void async_wait0() {
#if defined(__has_builtin)
#  if __has_builtin(__builtin_amdgcn_s_wait_asynccnt)
  __builtin_amdgcn_s_wait_asynccnt(0);
#  elif HAVE_ASYNC
  asm volatile("s_wait_asynccnt 0x0" ::: "memory");
#  endif
#endif
}

// ---------------------------------------------------------------------------
// WMMA fragment layout (wave32, V_WMMA_F32_16X16X32_BF16)
// B matrix (K=32 x N=16), per-lane v16bf:
//   lane L holds column N = L&15; hi = L>=16
//   elem i in [0,8):  K = 8*hi + i ; elem i in [8,16): K = 16 + 8*hi + (i-8)
// Both 512x512 B matrices are pre-swizzled in workspace so one fragment is a
// contiguous 32B chunk per lane (1 KB per 32x16 block).
// ---------------------------------------------------------------------------
__device__ __host__ inline size_t frag_pos(int k, int n) {
  int kb = k >> 5, kr = k & 31;
  int nb = n >> 4, nr = n & 15;
  int hi = (kr >> 3) & 1;
  int i  = (kr & 7) + ((kr >> 4) << 3);
  int lane = nr + (hi << 4);
  return ((size_t)(nb * 16 + kb) * 512) + (size_t)lane * 16 + i;
}

// A matrix (M=16 x K=32) fragment gather from row-major bf16 LDS tile
__device__ inline v16bf load_a_frag(const __bf16* lds, int row, int kb, int hi) {
  int k0 = kb * 32 + hi * 8;
  const v8bf* p0 = reinterpret_cast<const v8bf*>(lds + row * DDIM + k0);
  const v8bf* p1 = reinterpret_cast<const v8bf*>(lds + row * DDIM + k0 + 16);
  v8bf lo = *p0;
  v8bf hh = *p1;
  return __builtin_shufflevector(lo, hh, 0, 1, 2, 3, 4, 5, 6, 7,
                                         8, 9, 10, 11, 12, 13, 14, 15);
}

__device__ inline v8f wmma_bf16(v16bf a, v16bf b, v8f c) {
  return __builtin_amdgcn_wmma_f32_16x16x32_bf16(false, a, false, b,
                                                 (short)0, c, false, false);
}

// Stage one K-step's B stripe (32 fragments x 1 KB = 32 KB) from workspace
// (L2-resident) into LDS. 256 threads x 8 chunks x 16 B. Async DMA.
__device__ inline void stage_b(const __bf16* __restrict__ Bsrc, __bf16* lbuf,
                               int kb, int t) {
  #pragma unroll
  for (int i = 0; i < 8; ++i) {
    int c      = i * 256 + t;    // 16B chunk id, 0..2047
    int frag   = c >> 6;         // 0..31 (64 chunks per 1 KB fragment)
    int within = c & 63;
    const char* g = (const char*)Bsrc + (size_t)(frag * 16 + kb) * 1024 + within * 16;
    char*       l = (char*)lbuf + (size_t)c * 16;
#if HAVE_ASYNC
    __builtin_amdgcn_global_load_async_to_lds_b128((AS1 v4i*)g, (AS3 v4i*)l, 0, 0);
#else
    *(uint4*)l = *(const uint4*)g;
#endif
  }
}

// 16 WMMAs of one K-step with a 2-deep software pipeline of B-fragment LDS
// loads (keeps 2 fragment load-pairs in flight so ds latency hides under wmma)
__device__ inline void wmma_row(v16bf a_use, const __bf16* cur, int hcol,
                                int lane, v8f acc[16]) {
  const __bf16* bbase = cur + (size_t)(hcol * 16) * 512 + (size_t)lane * 16;
  v16bf bq0 = *(const v16bf*)(bbase);
  v16bf bq1 = *(const v16bf*)(bbase + 512);
  #pragma unroll
  for (int nb = 0; nb < 16; ++nb) {
    v16bf bn = bq1;
    if (nb < 14) bn = *(const v16bf*)(bbase + (size_t)(nb + 2) * 512);
    acc[nb] = wmma_bf16(a_use, bq0, acc[nb]);
    bq0 = bq1;
    bq1 = bn;
  }
}

// ---------------------------------------------------------------------------
// Prep: bf16 B-fragment images of cs^T (centers/lengthscale) and W_out^T;
// cnorm from the rounded values; also publishes 1/lengthscale.
// ---------------------------------------------------------------------------
__global__ __launch_bounds__(256) void rbf_prep(
    const float* __restrict__ centers, const float* __restrict__ ls,
    const float* __restrict__ W_out,
    __bf16* __restrict__ csb, __bf16* __restrict__ wb,
    float* __restrict__ cnorm, float* __restrict__ inv_ls) {
  __shared__ float red[256];
  const int n = blockIdx.x;
  const int t = threadIdx.x;
  float csq = 0.f;
  for (int k = t; k < DDIM; k += 256) {
    float inv = 1.f / ls[k];
    if (n == 0) inv_ls[k] = inv;
    float c = centers[(size_t)n * DDIM + k] * inv;
    __bf16 cb = (__bf16)c;
    csb[frag_pos(k, n)] = cb;
    float cf = (float)cb;
    csq += cf * cf;
    wb[frag_pos(k, n)] = (__bf16)W_out[(size_t)n * DDIM + k];
  }
  red[t] = csq;
  __syncthreads();
  for (int s = 128; s > 0; s >>= 1) {
    if (t < s) red[t] += red[t + s];
    __syncthreads();
  }
  if (t == 0) cnorm[n] = red[0];
}

// ---------------------------------------------------------------------------
// Fused main kernel. 8 waves/WG; wave (rb,hcol) owns 16 rows x 256 cols.
// B stripes double-buffered in LDS via async DMA, WMMAs fed from LDS with a
// software-pipelined fragment stream.
// ---------------------------------------------------------------------------
__global__ __launch_bounds__(256) void rbf_main(
    const float* __restrict__ x, const float* __restrict__ inv_ls,
    const float* __restrict__ b_out,
    const __bf16* __restrict__ csb, const __bf16* __restrict__ wb,
    const float* __restrict__ cnorm, float* __restrict__ out) {
  extern __shared__ __align__(16) char smem_raw[];
  __bf16* lds_mat  = (__bf16*)smem_raw;                             // 64x512 bf16 (reused)
  float*  lds_xnrm = (float*)(smem_raw + MBLK * DDIM * 2);          // 64 f32
  __bf16* lds_b0   = (__bf16*)(smem_raw + MBLK * DDIM * 2 + 256);   // 32 KB
  __bf16* lds_b1   = lds_b0 + 32 * 512;                             // 32 KB

  const int t    = threadIdx.x;
  const int lane = t & 31;
  const int wave = t >> 5;
  const int rb   = wave & 3;     // row stripe (16 rows)
  const int hcol = wave >> 2;    // column half (0/1)
  const int hi   = (lane >> 4) & 1;
  const int nlo  = lane & 15;
  const size_t m0 = (size_t)blockIdx.x * MBLK;

  // Kick off first B stripe DMA immediately; it runs under all of phase A.
  stage_b(csb, lds_b0, 0, t);

  if (t < MBLK) lds_xnrm[t] = 0.f;
  __syncthreads();

  // ---- Phase A: load/scale x tile, bf16 round, row norms of rounded values
  {
    const float* xblk = x + m0 * DDIM;
    float partial = 0.f;
    int curRow = 0;
    #pragma unroll 4
    for (int i = 0; i < (MBLK * DDIM) / 256; ++i) {
      int e   = i * 256 + t;
      int row = e >> 9;
      int col = e & (DDIM - 1);
      float v = xblk[e] * inv_ls[col];
      __bf16 xb = (__bf16)v;
      lds_mat[e] = xb;
      float xf = (float)xb;
      if (row != curRow) {
        atomicAdd(&lds_xnrm[curRow], partial);
        curRow = row;
        partial = 0.f;
      }
      partial += xf * xf;
    }
    atomicAdd(&lds_xnrm[curRow], partial);
  }
  async_wait0();
  __syncthreads();  // xs tile + norms + first B stripe ready

  const int rowA = rb * 16 + nlo;
  v8f acc[16];

  // ---- GEMM1: S = xs * cs^T (double-buffered LDS B stripes)
  #pragma unroll
  for (int i = 0; i < 16; ++i) acc[i] = (v8f){0.f, 0.f, 0.f, 0.f, 0.f, 0.f, 0.f, 0.f};

  v16bf a_cur = load_a_frag(lds_mat, rowA, 0, hi);
  #pragma unroll 1
  for (int kb = 0; kb < 16; ++kb) {
    __bf16* cur = (kb & 1) ? lds_b1 : lds_b0;
    __bf16* nxt = (kb & 1) ? lds_b0 : lds_b1;
    if (kb < 15) stage_b(csb, nxt, kb + 1, t);   // overlap DMA with WMMAs
    v16bf a_use = a_cur;
    if (kb < 15) a_cur = load_a_frag(lds_mat, rowA, kb + 1, hi);
    wmma_row(a_use, cur, hcol, lane, acc);
    async_wait0();
    __syncthreads();
  }

  // First W stripe DMA hides behind the exp epilogue.
  stage_b(wb, lds_b0, 0, t);

  // ---- Epilogue: sq = |x|^2 + |c|^2 - 2S ; K = exp(-0.5*max(sq,0)) -> LDS bf16
  {
    float xn[8];
    #pragma unroll
    for (int r = 0; r < 8; ++r) xn[r] = lds_xnrm[rb * 16 + hi * 8 + r];
    #pragma unroll
    for (int nb = 0; nb < 16; ++nb) {
      int n = (hcol * 16 + nb) * 16 + nlo;
      float cn = cnorm[n];
      #pragma unroll
      for (int r = 0; r < 8; ++r) {
        float sq = xn[r] + cn - 2.f * acc[nb][r];
        sq = fmaxf(sq, 0.f);
        float kv = __expf(-0.5f * sq);
        lds_mat[(rb * 16 + hi * 8 + r) * DDIM + n] = (__bf16)kv;
      }
    }
  }
  async_wait0();
  __syncthreads();  // K tile + first W stripe ready

  // ---- GEMM2: out = K * W_out^T + b
  #pragma unroll
  for (int i = 0; i < 16; ++i) acc[i] = (v8f){0.f, 0.f, 0.f, 0.f, 0.f, 0.f, 0.f, 0.f};

  a_cur = load_a_frag(lds_mat, rowA, 0, hi);
  #pragma unroll 1
  for (int kb = 0; kb < 16; ++kb) {
    __bf16* cur = (kb & 1) ? lds_b1 : lds_b0;
    __bf16* nxt = (kb & 1) ? lds_b0 : lds_b1;
    if (kb < 15) stage_b(wb, nxt, kb + 1, t);
    v16bf a_use = a_cur;
    if (kb < 15) a_cur = load_a_frag(lds_mat, rowA, kb + 1, hi);
    wmma_row(a_use, cur, hcol, lane, acc);
    async_wait0();
    __syncthreads();
  }

  #pragma unroll
  for (int nb = 0; nb < 16; ++nb) {
    int n = (hcol * 16 + nb) * 16 + nlo;
    float bo = b_out[n];
    #pragma unroll
    for (int r = 0; r < 8; ++r) {
      size_t mg = m0 + (size_t)(rb * 16 + hi * 8 + r);
      out[mg * DDIM + n] = acc[nb][r] + bo;
    }
  }
}

// ---------------------------------------------------------------------------
extern "C" void kernel_launch(void* const* d_in, const int* in_sizes, int n_in,
                              void* d_out, int out_size, void* d_ws, size_t ws_size,
                              hipStream_t stream) {
  const float* x       = (const float*)d_in[0];
  const float* centers = (const float*)d_in[1];
  const float* ls      = (const float*)d_in[2];
  const float* W_out   = (const float*)d_in[3];
  const float* b_out   = (const float*)d_in[4];
  float*       out     = (float*)d_out;

  char* ws = (char*)d_ws;
  __bf16* csb    = (__bf16*)ws;                        // 512 KB: cs^T fragments
  __bf16* wb     = (__bf16*)(ws + 512 * 1024);         // 512 KB: W^T fragments
  float*  cnorm  = (float*)(ws + 1024 * 1024);         // 2 KB
  float*  inv_ls = (float*)(ws + 1024 * 1024 + 2048);  // 2 KB

  rbf_prep<<<DDIM, 256, 0, stream>>>(centers, ls, W_out, csb, wb, cnorm, inv_ls);

  // 64 KB tile + 256 B norms + 2 x 32 KB B double-buffer = 131328 B
  size_t smem = (size_t)MBLK * DDIM * 2 + 256 + 2 * 32 * 512 * 2;
  rbf_main<<<NWG, 256, smem, stream>>>(x, inv_ls, b_out, csb, wb, cnorm, out);
}
